// CoarseMatching_26173530702387
// MI455X (gfx1250) — matrix operand order repaired and checked by reading, compiled, not run on hardware
//
#include <hip/hip_runtime.h>
#include <hip/hip_bf16.h>

typedef __attribute__((ext_vector_type(16))) __bf16 v16bf;
typedef __attribute__((ext_vector_type(8)))  __bf16 v8bf;
typedef __attribute__((ext_vector_type(8)))  float  v8f;
typedef __attribute__((ext_vector_type(4)))  float  v4f;

#define C_DIM 256
#define THR 0.2f
#define BORDER 2
#define KCH 8     // L-chunks for column-oriented passes
#define NT 5      // N-tiles per wave: 16*5 = 80 cols, 3600 = 80*45

// ---------------------------------------------------------------------------
// Pass 0: fp32 -> bf16 pre-conversion (done once; GEMM then streams bf16)
// ---------------------------------------------------------------------------
__global__ __launch_bounds__(256) void cm_cvt_bf16(
    const float* __restrict__ in, __bf16* __restrict__ out, long long total8)
{
  const long long i = (long long)blockIdx.x * 256 + threadIdx.x;
  if (i >= total8) return;
  const v4f* p = (const v4f*)in + i * 2;
  const v4f x = p[0], y = p[1];
  v8bf r;
  r[0] = (__bf16)x[0]; r[1] = (__bf16)x[1]; r[2] = (__bf16)x[2]; r[3] = (__bf16)x[3];
  r[4] = (__bf16)y[0]; r[5] = (__bf16)y[1]; r[6] = (__bf16)y[2]; r[7] = (__bf16)y[3];
  *((v8bf*)out + i) = r;
}

// ---------------------------------------------------------------------------
// Pass 1: sim[n,l,s] = dot(f0[n,l,:], f1[n,s,:]) / (C*TEMP) via bf16 WMMA.
// One wave computes a 16x80 strip (5 N-tiles share one A fragment).
// Manual double-buffering: fragments for K-step kk+32 are loaded before the
// WMMAs of step kk issue, so waits cover loads issued a full step earlier.
// Fully unrolled K-loop -> cur/nxt copies vanish via SSA renaming.
// L = 16*225, S = 80*45 -> exact tiling, EXEC all-ones for WMMA.
// ---------------------------------------------------------------------------
#define LOAD_A(kk, dst)                                                        \
  {                                                                            \
    const v8bf _lo = *(const v8bf*)(arow + (kk) + half * 8);                   \
    const v8bf _hi = *(const v8bf*)(arow + (kk) + half * 8 + 16);              \
    dst = __builtin_shufflevector(_lo, _hi, 0, 1, 2, 3, 4, 5, 6, 7, 8, 9, 10, \
                                  11, 12, 13, 14, 15);                         \
  }
#define LOAD_B(kk, t, dst)                                                     \
  {                                                                            \
    const __bf16* _bp = brow + (size_t)(t) * 16 * C_DIM + (kk) + half * 8;     \
    const v8bf _lo = *(const v8bf*)_bp;                                        \
    const v8bf _hi = *(const v8bf*)(_bp + 16);                                 \
    dst = __builtin_shufflevector(_lo, _hi, 0, 1, 2, 3, 4, 5, 6, 7, 8, 9, 10, \
                                  11, 12, 13, 14, 15);                         \
  }

__global__ __launch_bounds__(32) void cm_sim_gemm(
    const __bf16* __restrict__ a_bf, const __bf16* __restrict__ b_bf,
    float* __restrict__ sim, int L, int S)
{
  const int lane = threadIdx.x & 31;
  const int half = lane >> 4;      // 0: lanes 0-15, 1: lanes 16-31
  const int lm   = lane & 15;
  const int n  = blockIdx.z;
  const int m0 = blockIdx.y * 16;
  const int s0 = blockIdx.x * (16 * NT);

  const __bf16* __restrict__ arow = a_bf + ((size_t)n * L + m0 + lm) * C_DIM;
  const __bf16* __restrict__ brow = b_bf + ((size_t)n * S + s0 + lm) * C_DIM;

  v8f acc[NT] = {};

  // prologue: fragments for K-step 0
  v16bf a_cur;
  v16bf b_cur[NT];
  LOAD_A(0, a_cur);
#pragma unroll
  for (int t = 0; t < NT; ++t) LOAD_B(0, t, b_cur[t]);

#pragma unroll
  for (int kk = 0; kk < C_DIM; kk += 32) {
    v16bf a_nxt;
    v16bf b_nxt[NT];
    if (kk + 32 < C_DIM) {
      if (kk + 64 < C_DIM) {       // global_prefetch_b8 two steps ahead
        __builtin_prefetch(arow + kk + 64, 0, 1);
        __builtin_prefetch(brow + kk + 64, 0, 1);
      }
      LOAD_A(kk + 32, a_nxt);
#pragma unroll
      for (int t = 0; t < NT; ++t) LOAD_B(kk + 32, t, b_nxt[t]);
    }
#pragma unroll
    for (int t = 0; t < NT; ++t) {
      acc[t] = __builtin_amdgcn_wmma_f32_16x16x32_bf16(
          false, a_cur, false, b_cur[t], (short)0, acc[t], false, false);
    }
    a_cur = a_nxt;
#pragma unroll
    for (int t = 0; t < NT; ++t) b_cur[t] = b_nxt[t];
  }

  const float SCALE = 1.0f / (256.0f * 0.1f);   // 1/(C*TEMP)
  // D layout: VGPR j -> row m0 + half*8 + j, col s0 + 16*t + lm
  float* __restrict__ out = sim + ((size_t)n * L + m0 + half * 8) * S + s0 + lm;
#pragma unroll
  for (int j = 0; j < 8; ++j) {
#pragma unroll
    for (int t = 0; t < NT; ++t) out[(size_t)j * S + t * 16] = acc[t][j] * SCALE;
  }
}

// ---------------------------------------------------------------------------
// Pass 2a: per-row (n,l) online softmax stats over s -> max, 1/sum
// ---------------------------------------------------------------------------
__global__ __launch_bounds__(256) void cm_row_stats(
    const float* __restrict__ sim, float* __restrict__ rmax,
    float* __restrict__ rinv, int S)
{
  const int row = blockIdx.x;
  const int tid = threadIdx.x;
  const float* __restrict__ p = sim + (size_t)row * S;
  float m = -__builtin_inff(), sum = 0.f;
  for (int s = tid; s < S; s += 256) {
    float v = p[s];
    float nm = fmaxf(m, v);
    sum = sum * __expf(m - nm) + __expf(v - nm);
    m = nm;
  }
  __shared__ float sm[256], ss[256];
  sm[tid] = m; ss[tid] = sum;
  __syncthreads();
  for (int off = 128; off > 0; off >>= 1) {
    if (tid < off) {
      float m2 = sm[tid + off], s2 = ss[tid + off];
      float nm = fmaxf(sm[tid], m2);
      ss[tid] = ss[tid] * __expf(sm[tid] - nm) + s2 * __expf(m2 - nm);
      sm[tid] = nm;
    }
    __syncthreads();
  }
  if (tid == 0) { rmax[row] = sm[0]; rinv[row] = 1.0f / ss[0]; }
}

// ---------------------------------------------------------------------------
// Pass 2b: per-column (n,s) softmax stats over l, chunked; then combined
// ---------------------------------------------------------------------------
__global__ __launch_bounds__(256) void cm_col_stats_partial(
    const float* __restrict__ sim, float* __restrict__ pm,
    float* __restrict__ ps, int L, int S)
{
  const int s = blockIdx.x * 256 + threadIdx.x;
  const int chunk = blockIdx.y;
  const int n = blockIdx.z;
  if (s >= S) return;
  const int per = (L + KCH - 1) / KCH;
  const int l0 = chunk * per;
  const int l1 = (l0 + per < L) ? (l0 + per) : L;
  const float* __restrict__ p = sim + (size_t)n * L * S + s;
  float m = -__builtin_inff(), sum = 0.f;
  for (int l = l0; l < l1; ++l) {
    float v = p[(size_t)l * S];
    float nm = fmaxf(m, v);
    sum = sum * __expf(m - nm) + __expf(v - nm);
    m = nm;
  }
  const size_t o = ((size_t)n * KCH + chunk) * S + s;
  pm[o] = m; ps[o] = sum;
}

__global__ __launch_bounds__(256) void cm_col_combine(
    const float* __restrict__ pm, const float* __restrict__ ps,
    float* __restrict__ cmax, float* __restrict__ cinv, int S)
{
  const int s = blockIdx.x * 256 + threadIdx.x;
  const int n = blockIdx.y;
  if (s >= S) return;
  float m = -__builtin_inff(), sum = 0.f;
  for (int c = 0; c < KCH; ++c) {
    const size_t o = ((size_t)n * KCH + c) * S + s;
    float m2 = pm[o], s2 = ps[o];
    float nm = fmaxf(m, m2);
    sum = sum * __expf(m - nm) + s2 * __expf(m2 - nm);
    m = nm;
  }
  cmax[(size_t)n * S + s] = m;
  cinv[(size_t)n * S + s] = 1.0f / sum;
}

// ---------------------------------------------------------------------------
// Pass 3: conf = row_softmax * col_softmax (in place), column-oriented so the
// per-column conf max falls out for free (partial over L chunks).
// ---------------------------------------------------------------------------
__global__ __launch_bounds__(256) void cm_transform_col(
    float* __restrict__ conf, const float* __restrict__ rmax,
    const float* __restrict__ rinv, const float* __restrict__ cmax,
    const float* __restrict__ cinv, float* __restrict__ pcm, int L, int S)
{
  const int s = blockIdx.x * 256 + threadIdx.x;
  const int chunk = blockIdx.y;
  const int n = blockIdx.z;
  if (s >= S) return;
  const int per = (L + KCH - 1) / KCH;
  const int l0 = chunk * per;
  const int l1 = (l0 + per < L) ? (l0 + per) : L;
  const float cms = cmax[(size_t)n * S + s];
  const float ics = cinv[(size_t)n * S + s];
  float* __restrict__ p = conf + (size_t)n * L * S + s;
  const float* __restrict__ rm = rmax + (size_t)n * L;
  const float* __restrict__ ri = rinv + (size_t)n * L;
  float best = 0.f;
  for (int l = l0; l < l1; ++l) {
    const float v = p[(size_t)l * S];
    const float c = (__expf(v - rm[l]) * ri[l]) * (__expf(v - cms) * ics);
    p[(size_t)l * S] = c;
    best = fmaxf(best, c);
  }
  pcm[((size_t)n * KCH + chunk) * S + s] = best;
}

__global__ __launch_bounds__(256) void cm_colmax_combine(
    const float* __restrict__ pcm, float* __restrict__ ccfm, int S)
{
  const int s = blockIdx.x * 256 + threadIdx.x;
  const int n = blockIdx.y;
  if (s >= S) return;
  float m = 0.f;
  for (int c = 0; c < KCH; ++c) m = fmaxf(m, pcm[((size_t)n * KCH + c) * S + s]);
  ccfm[(size_t)n * S + s] = m;
}

// ---------------------------------------------------------------------------
// Pass 4: mutual-nearest extraction. Sweep 1 recomputes the row conf max
// (row is 14.4KB -> L0/L2-hot for sweep 2), sweep 2 finds first masked s.
// ---------------------------------------------------------------------------
__global__ __launch_bounds__(256) void cm_extract(
    const float* __restrict__ conf, const float* __restrict__ ccfm,
    const int* __restrict__ h0c, const int* __restrict__ w0c,
    const int* __restrict__ h1c, const int* __restrict__ w1c,
    float* __restrict__ mask_v, float* __restrict__ jids,
    float* __restrict__ mconf, int L, int S)
{
  const int row = blockIdx.x;
  const int n = row / L;
  const int l = row % L;
  const int tid = threadIdx.x;
  const int H0 = *h0c, W0 = *w0c, H1 = *h1c, W1 = *w1c;
  const int h0 = l / W0, w0 = l % W0;
  const bool vl = (h0 >= BORDER) && (h0 < H0 - BORDER) &&
                  (w0 >= BORDER) && (w0 < W0 - BORDER);
  const float* __restrict__ p = conf + (size_t)row * S;

  // sweep 1: row max of conf (exact fmaxf propagation -> == test is sound)
  float mloc = 0.f;
  for (int s = tid; s < S; s += 256) mloc = fmaxf(mloc, p[s]);
  __shared__ float sf[256];
  sf[tid] = mloc;
  __syncthreads();
  for (int off = 128; off > 0; off >>= 1) {
    if (tid < off) sf[tid] = fmaxf(sf[tid], sf[tid + off]);
    __syncthreads();
  }
  const float rm = sf[0];
  __syncthreads();

  // sweep 2: first s with full mask
  int bestS = S;
  if (vl) {
    const float* __restrict__ cm = ccfm + (size_t)n * S;
    for (int s = tid; s < S; s += 256) {
      const int h1 = s / W1, w1 = s % W1;
      const bool vs = (h1 >= BORDER) && (h1 < H1 - BORDER) &&
                      (w1 >= BORDER) && (w1 < W1 - BORDER);
      const float c = p[s];
      if (vs && (c > THR) && (c == rm) && (c == cm[s])) { bestS = s; break; }
    }
  }
  __shared__ int sb[256];
  sb[tid] = bestS;
  __syncthreads();
  for (int off = 128; off > 0; off >>= 1) {
    if (tid < off) sb[tid] = (sb[tid + off] < sb[tid]) ? sb[tid + off] : sb[tid];
    __syncthreads();
  }
  if (tid == 0) {
    const int j = sb[0];
    if (j < S) {
      mask_v[row] = 1.0f;
      jids[row]   = (float)j;
      mconf[row]  = p[j];
    } else {
      mask_v[row] = 0.0f;
      jids[row]   = 0.0f;
      mconf[row]  = 0.0f;
    }
  }
}

// ---------------------------------------------------------------------------
extern "C" void kernel_launch(void* const* d_in, const int* in_sizes, int n_in,
                              void* d_out, int out_size, void* d_ws, size_t ws_size,
                              hipStream_t stream) {
  const float* f0 = (const float*)d_in[0];
  const float* f1 = (const float*)d_in[1];
  const int* h0c = (const int*)d_in[2];
  const int* w0c = (const int*)d_in[3];
  const int* h1c = (const int*)d_in[4];
  const int* w1c = (const int*)d_in[5];

  const int L = 3600, S = 3600;                 // 60x60 coarse grids
  const int N = in_sizes[0] / (L * C_DIM);

  const size_t NL = (size_t)N * L, NS = (size_t)N * S;
  float* conf   = (float*)d_out;                 // [N,L,S]
  float* mask_v = conf + (size_t)N * L * S;      // [N,L]
  float* jids   = mask_v + NL;                   // [N,L]
  float* mconf  = jids + NL;                     // [N,L]

  // workspace: bf16 copies of the features, then float stats
  __bf16* abf = (__bf16*)d_ws;                   // N*L*C
  __bf16* bbf = abf + NL * C_DIM;                // N*S*C
  float* ws   = (float*)(bbf + NS * C_DIM);
  float* rmax = ws;                              // NL
  float* rinv = rmax + NL;                       // NL
  float* cmax = rinv + NL;                       // NS
  float* cinv = cmax + NS;                       // NS
  float* ccfm = cinv + NS;                       // NS
  float* pm   = ccfm + NS;                       // N*KCH*S (reused as pcm later)
  float* ps   = pm + (size_t)N * KCH * S;        // N*KCH*S

  const int sblk = (S + 255) / 256;
  const long long tot8a = (long long)NL * C_DIM / 8;
  const long long tot8b = (long long)NS * C_DIM / 8;

  // 0) bf16 pre-conversion of both feature tensors
  cm_cvt_bf16<<<(int)((tot8a + 255) / 256), 256, 0, stream>>>(f0, abf, tot8a);
  cm_cvt_bf16<<<(int)((tot8b + 255) / 256), 256, 0, stream>>>(f1, bbf, tot8b);
  // 1) sim GEMM via WMMA, written into the conf output region
  cm_sim_gemm<<<dim3(S / (16 * NT), L / 16, N), 32, 0, stream>>>(abf, bbf, conf, L, S);
  // 2) softmax stats
  cm_row_stats<<<(int)NL, 256, 0, stream>>>(conf, rmax, rinv, S);
  cm_col_stats_partial<<<dim3(sblk, KCH, N), 256, 0, stream>>>(conf, pm, ps, L, S);
  cm_col_combine<<<dim3(sblk, N), 256, 0, stream>>>(pm, ps, cmax, cinv, S);
  // 3) conf transform (column-oriented) + per-column conf max partials
  cm_transform_col<<<dim3(sblk, KCH, N), 256, 0, stream>>>(conf, rmax, rinv,
                                                           cmax, cinv, pm, L, S);
  cm_colmax_combine<<<dim3(sblk, N), 256, 0, stream>>>(pm, ccfm, S);
  // 4) mutual-nearest match extraction (recomputes row max in-kernel)
  cm_extract<<<(int)NL, 256, 0, stream>>>(conf, ccfm, h0c, w0c, h1c, w1c,
                                          mask_v, jids, mconf, L, S);
}
